// LogicRule_7017976562271
// MI455X (gfx1250) — matrix operand order repaired and checked by reading, not compiled, not run on hardware
//
#include <hip/hip_runtime.h>

typedef _Float16 v16h __attribute__((ext_vector_type(16)));
typedef float    v8f  __attribute__((ext_vector_type(8)));
typedef float    v4f  __attribute__((ext_vector_type(4)));

#define WMMA(A, Bm, C) __builtin_amdgcn_wmma_f32_16x16x32_f16( \
    false, (A), false, (Bm), (short)0, (C), false, false)

constexpr int Bn = 512, Wn = 1024, Ln = 64, Jn = 8, In = 32, On = 128;

// ---- dynamic LDS layout for k_attend ----
constexpr int OFF_WMHT = 0;                         // wm^T as f16 : [64][1024]
constexpr int OFF_SC   = OFF_WMHT + Ln * Wn * 2;    // scores/e    : [8][1024] f32
constexpr int OFF_W1   = OFF_SC   + Jn * Wn * 4;    // w1          : [8][64]  f32
constexpr int OFF_W1C  = OFF_W1   + Jn * Ln * 4;    // w1*c        : [8][64]  f32
constexpr int OFF_A    = OFF_W1C  + Jn * Ln * 4;    // A[j] (pad16): 16 f32
constexpr int OFF_INV  = OFF_A    + 16 * 4;         // 1/denom[j]  : 8 f32
constexpr int OFF_MQ   = OFF_INV  + 8 * 4;          // mq part[j]  : 8 f32
constexpr int OFF_RED  = OFF_MQ   + 8 * 4;          // best reduce : [4][16][16] f32
constexpr int LDS1_BYTES = OFF_RED + 4 * 256 * 4;   // = 172160 B (< 320 KB/WGP)

// ============================================================================
// Kernel 1: per-b attention.  scores -> softmax(-s) -> best[j,b,l], total_mq[b]
// wm is read from HBM exactly once; pass 2 runs entirely out of LDS (f16).
// ============================================================================
__global__ void __launch_bounds__(256) k_attend(
    const float* __restrict__ wm, const float* __restrict__ consts,
    const float* __restrict__ gammas, float* __restrict__ best,
    float* __restrict__ total_mq)
{
    extern __shared__ char smem[];
    _Float16* wmhT = (_Float16*)(smem + OFF_WMHT);
    float* sc   = (float*)(smem + OFF_SC);
    float* w1s  = (float*)(smem + OFF_W1);
    float* w1cs = (float*)(smem + OFF_W1C);
    float* Av   = (float*)(smem + OFF_A);
    float* invd = (float*)(smem + OFF_INV);
    float* mqp  = (float*)(smem + OFF_MQ);
    float* red  = (float*)(smem + OFF_RED);

    const int b    = blockIdx.x;
    const int tid  = threadIdx.x;
    const int wave = tid >> 5;
    const int lane = tid & 31;
    const int n    = lane & 15;   // column index of B/C fragments
    const int hi   = lane >> 4;   // half-wave selector

    // ---- prologue: w1 = 1-clip(g), w1*c, A[j] = sum w1*c*c ----
    for (int idx = tid; idx < Jn * Ln; idx += 256) {
        float g  = fminf(fmaxf(gammas[idx], 0.f), 1.f);
        float w1 = 1.f - g;
        w1s[idx]  = w1;
        w1cs[idx] = w1 * consts[idx];
    }
    __syncthreads();
    if (tid < 16) {
        float a = 0.f;
        if (tid < Jn)
            for (int l = 0; l < Ln; ++l) a += w1cs[tid * Ln + l] * consts[tid * Ln + l];
        Av[tid] = a;
    }
    __syncthreads();

    // ---- B fragments: BigB[k][j]:  k<64 -> -2*w1c[j][k],  k>=64 -> w1[j][k-64]
    v16h bf[4];
    #pragma unroll
    for (int kb = 0; kb < 4; ++kb) {
        #pragma unroll
        for (int t = 0; t < 16; ++t) {
            int k = 32 * kb + hi * 8 + t + (t & 8);
            float v = 0.f;
            if (n < Jn) v = (k < Ln) ? (-2.f * w1cs[n * Ln + k]) : w1s[n * Ln + (k - Ln)];
            bf[kb][t] = (_Float16)v;
        }
    }

    // ---- pass 1: stream wm, scores = A + [wm|wm^2] x BigB, stash wm^T(f16) ----
    for (int tile = 0; tile < 8; ++tile) {
        const int w0  = tile * 128 + wave * 16;
        const int row = w0 + n;
        const float* rp = wm + ((size_t)b * Wn + row) * Ln + hi * 8;
        __builtin_prefetch(rp + 128 * Ln, 0, 0);   // next tile -> global_prefetch_b8

        v4f q[8];
        #pragma unroll
        for (int c4 = 0; c4 < 4; ++c4) {
            q[2 * c4]     = *(const v4f*)(rp + c4 * 16);
            q[2 * c4 + 1] = *(const v4f*)(rp + c4 * 16 + 4);
        }
        v16h a0, a1, a2, a3;
        #pragma unroll
        for (int i = 0; i < 4; ++i) {
            a0[i]      = (_Float16)q[0][i];  a0[4 + i]  = (_Float16)q[1][i];
            a0[8 + i]  = (_Float16)q[2][i];  a0[12 + i] = (_Float16)q[3][i];
            a1[i]      = (_Float16)q[4][i];  a1[4 + i]  = (_Float16)q[5][i];
            a1[8 + i]  = (_Float16)q[6][i];  a1[12 + i] = (_Float16)q[7][i];
            a2[i]      = (_Float16)(q[0][i] * q[0][i]);
            a2[4 + i]  = (_Float16)(q[1][i] * q[1][i]);
            a2[8 + i]  = (_Float16)(q[2][i] * q[2][i]);
            a2[12 + i] = (_Float16)(q[3][i] * q[3][i]);
            a3[i]      = (_Float16)(q[4][i] * q[4][i]);
            a3[4 + i]  = (_Float16)(q[5][i] * q[5][i]);
            a3[8 + i]  = (_Float16)(q[6][i] * q[6][i]);
            a3[12 + i] = (_Float16)(q[7][i] * q[7][i]);
        }
        // stash transposed f16 copy for pass 2 (B-fragments read rows of wm^T)
        #pragma unroll
        for (int i = 0; i < 8; ++i) {
            wmhT[(hi * 8 + i) * Wn + row]        = a0[i];
            wmhT[(16 + hi * 8 + i) * Wn + row]   = a0[8 + i];
            wmhT[(32 + hi * 8 + i) * Wn + row]   = a1[i];
            wmhT[(48 + hi * 8 + i) * Wn + row]   = a1[8 + i];
        }

        v8f c;
        const float aj = Av[n];
        #pragma unroll
        for (int r = 0; r < 8; ++r) c[r] = aj;
        c = WMMA(a0, bf[0], c);
        c = WMMA(a1, bf[1], c);
        c = WMMA(a2, bf[2], c);
        c = WMMA(a3, bf[3], c);

        if (n < Jn) {  // only j<8 columns are real; don't overflow sc region
            float* sp = sc + n * Wn + w0 + 8 * hi;
            v4f lo = {c[0], c[1], c[2], c[3]};
            v4f hi4 = {c[4], c[5], c[6], c[7]};
            *(v4f*)sp       = lo;
            *(v4f*)(sp + 4) = hi4;
        }
    }
    __syncthreads();

    // ---- softmax(-s) stats: one wave per j ----
    {
        const int j = wave;                      // 8 waves == Jn
        float mn = 3.0e38f;
        for (int w = lane; w < Wn; w += 32) mn = fminf(mn, sc[j * Wn + w]);
        #pragma unroll
        for (int o = 16; o > 0; o >>= 1) mn = fminf(mn, __shfl_xor(mn, o));
        float se = 0.f, ses = 0.f;
        for (int w = lane; w < Wn; w += 32) {
            float s = sc[j * Wn + w];
            float e = __expf(mn - s);            // exp(-s - max(-s))
            sc[j * Wn + w] = e;                  // overwrite with unnormalized attn
            se += e; ses += e * s;
        }
        #pragma unroll
        for (int o = 16; o > 0; o >>= 1) { se += __shfl_xor(se, o); ses += __shfl_xor(ses, o); }
        if (lane == 0) { invd[j] = 1.f / se; mqp[j] = ses / se; }
    }
    __syncthreads();
    if (tid == 0) {
        float t = 0.f;
        for (int j = 0; j < Jn; ++j) t += mqp[j];
        total_mq[b] = t;
    }

    // ---- pass 2: best[j,l] = (sum_w e[j,w]*wm[w,l]) * invden[j], all from LDS ----
    const int ltile = wave & 3, l0 = ltile * 16, khalf = wave >> 2;
    const int jr = n & 7;                        // clamp padded rows 8..15 -> dupes
    v8f acc;
    #pragma unroll
    for (int r = 0; r < 8; ++r) acc[r] = 0.f;
    for (int kw = khalf * 16; kw < khalf * 16 + 16; ++kw) {
        const float* ep = sc + jr * Wn + 32 * kw + hi * 8;
        v4f e0 = *(const v4f*)ep,        e1 = *(const v4f*)(ep + 4);
        v4f e2 = *(const v4f*)(ep + 16), e3 = *(const v4f*)(ep + 20);
        v16h ae;
        #pragma unroll
        for (int i = 0; i < 4; ++i) {
            ae[i]      = (_Float16)e0[i];  ae[4 + i]  = (_Float16)e1[i];
            ae[8 + i]  = (_Float16)e2[i];  ae[12 + i] = (_Float16)e3[i];
        }
        const _Float16* wp = wmhT + (l0 + n) * Wn + 32 * kw + hi * 8;
        v16h bw;
        #pragma unroll
        for (int t = 0; t < 16; ++t) bw[t] = wp[t + (t & 8)];
        acc = WMMA(ae, bw, acc);
    }
    if (khalf == 1) {
        #pragma unroll
        for (int r = 0; r < 8; ++r) red[ltile * 256 + (8 * hi + r) * 16 + n] = acc[r];
    }
    __syncthreads();
    if (khalf == 0 && hi == 0) {
        #pragma unroll
        for (int r = 0; r < 8; ++r) {   // m = r = j  (rows 8..15 are dup-padding)
            float v = (acc[r] + red[ltile * 256 + r * 16 + n]) * invd[r];
            best[((size_t)r * Bn + b) * Ln + l0 + n] = v;
        }
    }
}

// ============================================================================
// Kernel 2: body + slot-selector softmax + head, 16 b-rows per block.
// ============================================================================
__global__ void __launch_bounds__(256) k_head(
    const float* __restrict__ best,   const float* __restrict__ total_mq,
    const float* __restrict__ gammas, const float* __restrict__ body_w,
    const float* __restrict__ body_b, const float* __restrict__ slot_w,
    const float* __restrict__ slot_b, const float* __restrict__ head_w,
    const float* __restrict__ head_b, float* __restrict__ out)
{
    __shared__ float bestl[16 * 64];   // best[j] tile for current j
    __shared__ float cv[16 * 32];      // captured_vars[b][i]
    __shared__ float conf[16];
    __shared__ float gavg[8];

    const int tid = threadIdx.x, wave = tid >> 5, lane = tid & 31;
    const int n = lane & 15, hi = lane >> 4;
    const int b0 = blockIdx.x * 16;

    if (tid < 16) conf[tid] = __expf(-total_mq[b0 + tid]);
    if (tid >= 32 && tid < 40) {
        int j = tid - 32; float s = 0.f;
        for (int l = 0; l < Ln; ++l) s += fminf(fmaxf(gammas[j * Ln + l], 0.f), 1.f);
        gavg[j] = s * (1.f / (float)Ln);
    }
    for (int idx = tid; idx < 16 * 32; idx += 256) cv[idx] = 0.f;
    __syncthreads();

    float accA[8], accB[8], cb[8];
    #pragma unroll
    for (int r = 0; r < 8; ++r) { accA[r] = 0.f; accB[r] = 0.f; cb[r] = 0.f; }

    for (int j = 0; j < Jn; ++j) {
        for (int idx = tid; idx < 16 * 64; idx += 256)
            bestl[idx] = best[((size_t)j * Bn + b0 + (idx >> 6)) * Ln + (idx & 63)];
        __syncthreads();

        // A fragments: rows = b (m = n), K = l (64 -> 2 frags). Reused 128x.
        v16h af0, af1;
        #pragma unroll
        for (int i = 0; i < 8; ++i) {
            af0[i]     = (_Float16)bestl[n * 64 + hi * 16 + i];
            af0[8 + i] = (_Float16)bestl[n * 64 + hi * 16 + 8 + i];
            af1[i]     = (_Float16)bestl[n * 64 + 32 + hi * 16 + i];
            af1[8 + i] = (_Float16)bestl[n * 64 + 32 + hi * 16 + 8 + i];
        }

        // ---- slot selector: 64 groups of 32 k (= one l each), 8 per wave ----
        for (int grp = wave; grp < 64; grp += 8) {
            const int k0 = grp * 32;
            v8f c0, c1;
            {
                float bias0 = slot_b[j * 2048 + k0 + n];
                float bias1 = slot_b[j * 2048 + k0 + 16 + n];
                #pragma unroll
                for (int r = 0; r < 8; ++r) { c0[r] = bias0; c1[r] = bias1; }
            }
            {   // k-tile 0 : i = n
                const float* sp = slot_w + ((size_t)j * 2048 + k0 + n) * 64 + hi * 16;
                v16h bs0, bs1;
                #pragma unroll
                for (int t = 0; t < 16; ++t) { bs0[t] = (_Float16)sp[t]; bs1[t] = (_Float16)sp[32 + t]; }
                c0 = WMMA(af0, bs0, c0);
                c0 = WMMA(af1, bs1, c0);
            }
            {   // k-tile 1 : i = 16 + n
                const float* sp = slot_w + ((size_t)j * 2048 + k0 + 16 + n) * 64 + hi * 16;
                v16h bs0, bs1;
                #pragma unroll
                for (int t = 0; t < 16; ++t) { bs0[t] = (_Float16)sp[t]; bs1[t] = (_Float16)sp[32 + t]; }
                c1 = WMMA(af0, bs0, c1);
                c1 = WMMA(af1, bs1, c1);
            }
            // softmax over i (32 values: 16 lanes x 2 regs) per (b-row r, half hi)
            #pragma unroll
            for (int r = 0; r < 8; ++r) {
                float bv = bestl[(8 * hi + r) * 64 + grp];   // best[j, b, l=grp]
                float mx = fmaxf(c0[r], c1[r]);
                #pragma unroll
                for (int o = 8; o > 0; o >>= 1) mx = fmaxf(mx, __shfl_xor(mx, o));
                float e0 = __expf(c0[r] - mx), e1 = __expf(c1[r] - mx);
                float s = e0 + e1;
                #pragma unroll
                for (int o = 8; o > 0; o >>= 1) s += __shfl_xor(s, o);
                float inv = 1.f / s;
                accA[r] += e0 * inv * bv;
                accB[r] += e1 * inv * bv;
            }
        }

        // ---- body: captured_vars += g_avg[j] * (best . body_w[j]) (waves 0,1) ----
        if (wave < 2) {
            const int i0 = wave * 16;
            const float* bp = body_w + ((size_t)j * In + i0 + n) * 64 + hi * 16;
            v16h bb0, bb1;
            #pragma unroll
            for (int t = 0; t < 16; ++t) { bb0[t] = (_Float16)bp[t]; bb1[t] = (_Float16)bp[32 + t]; }
            v8f c;
            #pragma unroll
            for (int r = 0; r < 8; ++r) c[r] = 0.f;
            c = WMMA(af0, bb0, c);
            c = WMMA(af1, bb1, c);
            const float ga = gavg[j];
            #pragma unroll
            for (int r = 0; r < 8; ++r) cb[r] += ga * c[r];
        }
        __syncthreads();   // bestl reused next j
    }

    // ---- fold per-wave partials into cv ----
    #pragma unroll
    for (int r = 0; r < 8; ++r) {
        atomicAdd(&cv[(8 * hi + r) * 32 + n],      accA[r]);   // ds_add_f32
        atomicAdd(&cv[(8 * hi + r) * 32 + 16 + n], accB[r]);
    }
    if (wave < 2) {
        const int i0 = wave * 16;
        float bbias = 0.f;
        for (int j = 0; j < Jn; ++j) bbias += gavg[j] * body_b[j * In + i0 + n];
        #pragma unroll
        for (int r = 0; r < 8; ++r) atomicAdd(&cv[(8 * hi + r) * 32 + i0 + n], cb[r] + bbias);
    }
    __syncthreads();

    // ---- head: out[b,o] = conf[b] * (cv . head_w^T + head_b), one o-tile/wave ----
    {
        const int o0 = wave * 16;
        v16h acv, bh;
        #pragma unroll
        for (int i = 0; i < 8; ++i) {
            acv[i]     = (_Float16)cv[n * 32 + hi * 16 + i];
            acv[8 + i] = (_Float16)cv[n * 32 + hi * 16 + 8 + i];
        }
        const float* hp = head_w + (o0 + n) * In + hi * 16;
        #pragma unroll
        for (int t = 0; t < 16; ++t) bh[t] = (_Float16)hp[t];
        v8f c;
        const float hb = head_b[o0 + n];
        #pragma unroll
        for (int r = 0; r < 8; ++r) c[r] = hb;
        c = WMMA(acv, bh, c);
        #pragma unroll
        for (int r = 0; r < 8; ++r) {
            const int m = 8 * hi + r;
            out[(size_t)(b0 + m) * On + o0 + n] = conf[m] * c[r];
        }
    }
}

extern "C" void kernel_launch(void* const* d_in, const int* in_sizes, int n_in,
                              void* d_out, int out_size, void* d_ws, size_t ws_size,
                              hipStream_t stream)
{
    const float* wm     = (const float*)d_in[0];
    const float* consts = (const float*)d_in[1];
    const float* gammas = (const float*)d_in[2];
    const float* body_w = (const float*)d_in[3];
    const float* body_b = (const float*)d_in[4];
    const float* slot_w = (const float*)d_in[5];
    const float* slot_b = (const float*)d_in[6];
    const float* head_w = (const float*)d_in[7];
    const float* head_b = (const float*)d_in[8];
    float* out = (float*)d_out;

    float* best = (float*)d_ws;                   // J*B*L f32 = 1 MB
    float* mq   = best + (size_t)Jn * Bn * Ln;    // B f32

    k_attend<<<dim3(Bn), dim3(256), LDS1_BYTES, stream>>>(wm, consts, gammas, best, mq);
    k_head<<<dim3(Bn / 16), dim3(256), 0, stream>>>(best, mq, gammas, body_w, body_b,
                                                    slot_w, slot_b, head_w, head_b, out);
}
